// GraphAdapter_62732292326152
// MI455X (gfx1250) — matrix operand order repaired and checked by reading, not compiled, not run on hardware
//
#include <hip/hip_runtime.h>
#include <hip/hip_bf16.h>
#include <stdint.h>

#define NN    30522
#define NPAD  30528            // 16-aligned row count for WMMA M tiles
#define HID   768
#define EDG   488352
#define BQ    4
#define SQ    512
#define TQ    (BQ*SQ)          // 2048 tokens
#define LQ    12
#define CR    101              // C+1 codebook rows
#define HH    (HID*HID)
#define LDSP  776              // padded LDS row stride (bf16) -> conflict-free ds_load_b128
#define SHB   (64*LDSP*2)      // dynamic LDS bytes for one 64-col B strip
#define EPB   4                // edges per scatter block

typedef __bf16 bf16;
typedef __bf16 v16bf __attribute__((ext_vector_type(16)));
typedef __bf16 v8bf  __attribute__((ext_vector_type(8)));
typedef float  v8f   __attribute__((ext_vector_type(8)));
typedef int    v4i   __attribute__((vector_size(16)));
typedef __attribute__((address_space(1))) v4i* gv4i_p;   // global int4*
typedef __attribute__((address_space(3))) v4i* lv4i_p;   // LDS int4*

#if __has_builtin(__builtin_amdgcn_global_load_async_to_lds_b128)
#define HAS_ASYNC_LDS 1
#endif

#define WMMA_BF16(a, b, c) \
  __builtin_amdgcn_wmma_f32_16x16x32_bf16(false, (a), false, (b), (short)0, (c), false, false)

// A 16x32 bf16 fragment: half 0 -> K 0..15, half 1 -> K 16..31 (matches B)
__device__ __forceinline__ v16bf ld_afrag(const bf16* Ap, int k0, int hf) {
  v8bf lo = *(const v8bf*)(Ap + k0 + hf * 16);
  v8bf hi = *(const v8bf*)(Ap + k0 + hf * 16 + 8);
  return __builtin_shufflevector(lo, hi, 0,1,2,3,4,5,6,7,8,9,10,11,12,13,14,15);
}
// B 32x16 fragment: contiguous 16 bf16 of one Bt row (column of B)
__device__ __forceinline__ v16bf ld_bfrag(const bf16* p) {
  v8bf lo = *(const v8bf*)p;
  v8bf hi = *(const v8bf*)(p + 8);
  return __builtin_shufflevector(lo, hi, 0,1,2,3,4,5,6,7,8,9,10,11,12,13,14,15);
}
// stage one 64-col B strip (64 x K bf16, padded LDS rows) - async copy unit
__device__ __forceinline__ void stage_bstrip(bf16* ldsB, const bf16* Bs, int K) {
#ifdef HAS_ASYNC_LDS
  for (int q = threadIdx.x; q < 64 * (K / 8); q += blockDim.x) {
    int row = q / (K / 8), cc = q - row * (K / 8);
    __builtin_amdgcn_global_load_async_to_lds_b128(
        (gv4i_p)(Bs + (size_t)row * K + cc * 8),
        (lv4i_p)(ldsB + row * LDSP + cc * 8), 0, 0);
  }
#if __has_builtin(__builtin_amdgcn_s_wait_asynccnt)
  __builtin_amdgcn_s_wait_asynccnt(0);
#else
  asm volatile("s_wait_asynccnt 0" ::: "memory");
#endif
#else
  for (int q = threadIdx.x; q < 64 * (K / 8); q += blockDim.x) {
    int row = q / (K / 8), cc = q - row * (K / 8);
    *(float4*)(ldsB + row * LDSP + cc * 8) = *(const float4*)(Bs + (size_t)row * K + cc * 8);
  }
#endif
  __syncthreads();
}

// ------------------------------------------------------------------
// utility kernels
// ------------------------------------------------------------------
__global__ void k_zero_f32(float* p, long n) {
  long i = (long)blockIdx.x * blockDim.x + threadIdx.x;
  if (i < n) p[i] = 0.f;
}
__global__ void k_zero_bf16(bf16* p, long n) {
  long i = (long)blockIdx.x * blockDim.x + threadIdx.x;
  if (i < n) p[i] = (bf16)0.f;
}
__global__ void k_stage_pad(bf16* dst, const float* __restrict__ src, long nvalid, long ntot) {
  long i = (long)blockIdx.x * blockDim.x + threadIdx.x;
  if (i < ntot) dst[i] = (i < nvalid) ? (bf16)src[i] : (bf16)0.f;
}
__global__ void k_stage_direct(bf16* dst, const float* __restrict__ src, long n) {
  long i = (long)blockIdx.x * blockDim.x + threadIdx.x;
  if (i < n) dst[i] = (bf16)src[i];
}
// transpose-stage: Bt[n*H+k] = W[k*H+n]  (GCN weights stored [in][out])
__global__ void k_stage_T(bf16* dst, const float* __restrict__ src) {
  long base = (long)blockIdx.z * HH;
  int idx = blockIdx.x * blockDim.x + threadIdx.x;     // < HH
  int k = idx / HID, n = idx - k * HID;
  dst[base + (long)n * HID + k] = (bf16)src[base + idx];
}
// ------------------------------------------------------------------
// GCN normalization
// ------------------------------------------------------------------
__global__ void k_deg(float* deg, const int* __restrict__ dst, const float* __restrict__ ew, int e) {
  int i = blockIdx.x * blockDim.x + threadIdx.x;
  if (i < e) atomicAdd(&deg[dst[i]], ew[i]);
}
__global__ void k_dinv(float* d, int n) {
  int i = blockIdx.x * blockDim.x + threadIdx.x;
  if (i < n) d[i] = rsqrtf(d[i] + 1.0f);
}
// ------------------------------------------------------------------
// Register-blocked WMMA GEMM: C[M,768] = A[M,768] * B, Bt = B^T (768x768)
// block = 8 waves; block stages one 64-col B strip in LDS (padded rows),
// each wave computes a 32x64 tile (2 M-tiles x 4 N-tiles, 8 WMMA / K-step)
// ------------------------------------------------------------------
__global__ void k_gemm2(const bf16* __restrict__ A, const bf16* __restrict__ Bt,
                        float* __restrict__ C, int mpairs) {
  extern __shared__ bf16 ldsB[];
  const int K = HID, N = HID;
  int strip = blockIdx.y;                               // 0..11 (64 cols each)
  stage_bstrip(ldsB, Bt + (size_t)strip * 64 * K, K);
  int wv = threadIdx.x >> 5, lane = threadIdx.x & 31, hf = lane >> 4, mr = lane & 15;
  int pair = blockIdx.x * 8 + wv;
  if (pair >= mpairs) return;                           // after barrier: safe
  const bf16* Ap0 = A + ((long)pair * 32 + mr) * K;
  const bf16* Ap1 = Ap0 + (long)16 * K;
  v8f acc[2][4];
#pragma unroll
  for (int m = 0; m < 2; m++)
#pragma unroll
    for (int j = 0; j < 4; j++) acc[m][j] = (v8f){};
  for (int k0 = 0; k0 < K; k0 += 32) {
    // preload all fragments into distinct registers, then 8 back-to-back WMMAs
    v16bf a0 = ld_afrag(Ap0, k0, hf);
    v16bf a1 = ld_afrag(Ap1, k0, hf);
    const bf16* bp = ldsB + mr * LDSP + k0 + hf * 16;
    v16bf b0 = ld_bfrag(bp);
    v16bf b1 = ld_bfrag(bp + 16 * LDSP);
    v16bf b2 = ld_bfrag(bp + 32 * LDSP);
    v16bf b3 = ld_bfrag(bp + 48 * LDSP);
    acc[0][0] = WMMA_BF16(a0, b0, acc[0][0]);
    acc[1][0] = WMMA_BF16(a1, b0, acc[1][0]);
    acc[0][1] = WMMA_BF16(a0, b1, acc[0][1]);
    acc[1][1] = WMMA_BF16(a1, b1, acc[1][1]);
    acc[0][2] = WMMA_BF16(a0, b2, acc[0][2]);
    acc[1][2] = WMMA_BF16(a1, b2, acc[1][2]);
    acc[0][3] = WMMA_BF16(a0, b3, acc[0][3]);
    acc[1][3] = WMMA_BF16(a1, b3, acc[1][3]);
  }
  long rbase = (long)pair * 32 + hf * 8;
  int cbase = strip * 64 + mr;
#pragma unroll
  for (int m = 0; m < 2; m++)
#pragma unroll
    for (int j = 0; j < 4; j++)
#pragma unroll
      for (int r = 0; r < 8; r++)
        C[(rbase + m * 16 + r) * (long)N + cbase + j * 16] = acc[m][j][r];
}
// ------------------------------------------------------------------
// edge scatter with lookahead prefetch: agg[dst] += nrm * hW[src]
// ------------------------------------------------------------------
__global__ void k_scatter(float* __restrict__ agg, const float* __restrict__ h,
                          const int* __restrict__ src, const int* __restrict__ dst,
                          const float* __restrict__ ew, const float* __restrict__ dinv) {
  int e0 = blockIdx.x * EPB;
  int c = threadIdx.x;                                  // 192 threads * 4 ch = 768
#pragma unroll
  for (int i = 0; i < EPB; i++) {
    int e = e0 + i;
    if (i + 1 < EPB) {                                  // prefetch next edge's source row
      int sn = src[e + 1];
      __builtin_prefetch(h + (long)sn * HID + c * 4, 0, 0);
    }
    int s = src[e], d = dst[e];
    float nrm = dinv[s] * ew[e] * dinv[d];
    float4 hv = *(const float4*)(h + (long)s * HID + c * 4);
    float* ad = agg + (long)d * HID + c * 4;
    atomicAdd(&ad[0], nrm * hv.x);
    atomicAdd(&ad[1], nrm * hv.y);
    atomicAdd(&ad[2], nrm * hv.z);
    atomicAdd(&ad[3], nrm * hv.w);
  }
}
// h_new = relu(agg + hW*dinv^2 + b); write f32 in place + bf16 stage
__global__ void k_finalize(float* __restrict__ hW, const float* __restrict__ agg,
                           const float* __restrict__ dinv, const float* __restrict__ bias,
                           bf16* __restrict__ hb) {
  int row = blockIdx.y;
  int c = blockIdx.x * blockDim.x + threadIdx.x;
  float di = dinv[row];
  long o = (long)row * HID + c;
  float v = agg[o] + hW[o] * (di * di) + bias[c];
  v = fmaxf(v, 0.f);
  hW[o] = v;
  hb[o] = (bf16)v;
}
__global__ void k_gather(float* __restrict__ acc, const float* __restrict__ h,
                         const int* __restrict__ ids) {
  int t = blockIdx.y;
  int c = blockIdx.x * blockDim.x + threadIdx.x;
  int id = ids[t];
  float v = (id < NN) ? h[(long)id * HID + c] : 0.f;
  acc[(long)t * HID + c] += v;
}
__global__ void k_tok(bf16* __restrict__ dst, const float* __restrict__ src, float sc, long n) {
  long i = (long)blockIdx.x * blockDim.x + threadIdx.x;
  if (i < n) dst[i] = (bf16)(src[i] * sc);
}
// pn_small[r,c] = sum_k pn_graph[r,k] * (cb[k,c] + pn_b[c]) ; cb (94MB) lives in L2
__global__ void k_pn_small(float* __restrict__ pns, const float* __restrict__ g,
                           const float* __restrict__ cb, const float* __restrict__ pnb) {
  int row = blockIdx.y;
  int c = blockIdx.x * blockDim.x + threadIdx.x;
  float bv = pnb[c], acc = 0.f;
  const float* gr = g + (long)row * NN;
#pragma unroll 4
  for (int k = 0; k < NN; k++) acc += gr[k] * (cb[(long)k * HID + c] + bv);
  pns[row * HID + c] = acc;
}
__global__ void k_pn_gather(bf16* __restrict__ tok, const float* __restrict__ pns,
                            const int* __restrict__ cids) {
  int t = blockIdx.y;
  int c = blockIdx.x * blockDim.x + threadIdx.x;
  tok[(long)t * HID + c] = (bf16)pns[(long)cids[t] * HID + c];
}
// ------------------------------------------------------------------
// zero-conv: same blocked WMMA structure, fused bias/mask/strided store
// grid = (8 mpair-blocks, 12 strips, L layers)
// ------------------------------------------------------------------
__global__ void k_zeroconv2(const bf16* __restrict__ tok, const bf16* __restrict__ Wt,
                            const float* __restrict__ bias, const int* __restrict__ mask,
                            float* __restrict__ out) {
  extern __shared__ bf16 ldsB[];
  const int K = HID;
  int l = blockIdx.z, strip = blockIdx.y;
  stage_bstrip(ldsB, Wt + (size_t)l * HH + (size_t)strip * 64 * K, K);
  int wv = threadIdx.x >> 5, lane = threadIdx.x & 31, hf = lane >> 4, mr = lane & 15;
  int pair = blockIdx.x * 8 + wv;                       // < 64 (TQ/32)
  const bf16* Ap0 = tok + ((long)pair * 32 + mr) * K;
  const bf16* Ap1 = Ap0 + (long)16 * K;
  v8f acc[2][4];
#pragma unroll
  for (int m = 0; m < 2; m++)
#pragma unroll
    for (int j = 0; j < 4; j++) acc[m][j] = (v8f){};
  for (int k0 = 0; k0 < K; k0 += 32) {
    v16bf a0 = ld_afrag(Ap0, k0, hf);
    v16bf a1 = ld_afrag(Ap1, k0, hf);
    const bf16* bp = ldsB + mr * LDSP + k0 + hf * 16;
    v16bf b0 = ld_bfrag(bp);
    v16bf b1 = ld_bfrag(bp + 16 * LDSP);
    v16bf b2 = ld_bfrag(bp + 32 * LDSP);
    v16bf b3 = ld_bfrag(bp + 48 * LDSP);
    acc[0][0] = WMMA_BF16(a0, b0, acc[0][0]);
    acc[1][0] = WMMA_BF16(a1, b0, acc[1][0]);
    acc[0][1] = WMMA_BF16(a0, b1, acc[0][1]);
    acc[1][1] = WMMA_BF16(a1, b1, acc[1][1]);
    acc[0][2] = WMMA_BF16(a0, b2, acc[0][2]);
    acc[1][2] = WMMA_BF16(a1, b2, acc[1][2]);
    acc[0][3] = WMMA_BF16(a0, b3, acc[0][3]);
    acc[1][3] = WMMA_BF16(a1, b3, acc[1][3]);
  }
#pragma unroll
  for (int m = 0; m < 2; m++)
#pragma unroll
    for (int j = 0; j < 4; j++) {
      int col = strip * 64 + j * 16 + mr;
      float bv = bias[l * HID + col];
#pragma unroll
      for (int r = 0; r < 8; r++) {
        int t = pair * 32 + m * 16 + hf * 8 + r;
        float mv = mask ? (mask[t] ? 0.f : 1.f) : 1.f;
        int b = t >> 9, s2 = t & (SQ - 1);
        out[(((long)l * BQ + b) * (SQ + 1) + (s2 + 1)) * HID + col] = (acc[m][j][r] + bv) * mv;
      }
    }
}
// zero the s==0 pad rows of all three output tensors
__global__ void k_outpad(float* out) {
  int i = blockIdx.x * blockDim.x + threadIdx.x;   // 3*L*B*H = 110592
  if (i >= 3 * LQ * BQ * HID) return;
  int c = i % HID; int rest = i / HID;
  int b = rest % BQ; rest /= BQ;
  int l = rest % LQ; int tns = rest / LQ;
  long OUTSZ = (long)LQ * BQ * (SQ + 1) * HID;
  out[tns * OUTSZ + ((long)l * BQ + b) * (SQ + 1) * HID + c] = 0.f;
}

// ------------------------------------------------------------------
extern "C" void kernel_launch(void* const* d_in, const int* in_sizes, int n_in,
                              void* d_out, int out_size, void* d_ws, size_t ws_size,
                              hipStream_t stream) {
  (void)in_sizes; (void)n_in; (void)out_size; (void)ws_size;
  const float* word       = (const float*)d_in[0];
  const int*   input_ids  = (const int*)d_in[1];
  const int*   class_ids  = (const int*)d_in[2];
  const int*   mask       = (const int*)d_in[3];
  const int*   co_ei      = (const int*)d_in[4];
  const float* co_ew      = (const float*)d_in[5];
  const int*   re_ei      = (const int*)d_in[6];
  const float* re_ew      = (const float*)d_in[7];
  const float* pn_graph   = (const float*)d_in[8];
  const float* co_W       = (const float*)d_in[9];
  const float* co_b       = (const float*)d_in[10];
  const float* re_W       = (const float*)d_in[11];
  const float* re_b       = (const float*)d_in[12];
  const float* pn_W       = (const float*)d_in[13];
  const float* pn_b       = (const float*)d_in[14];
  const float* zero_w     = (const float*)d_in[15];
  const float* zero_b     = (const float*)d_in[16];
  const float* zero_w_pos = (const float*)d_in[17];
  const float* zero_b_pos = (const float*)d_in[18];

  uint8_t* base = (uint8_t*)d_ws;
  size_t off = 0;
  auto carve = [&](size_t bytes) -> void* {
    void* p = base + off;
    off += bytes; off = (off + 255) & ~(size_t)255;
    return p;
  };
  bf16*  wordb = (bf16*) carve((size_t)NPAD * HID * 2);
  bf16*  hb    = (bf16*) carve((size_t)NPAD * HID * 2);
  float* hW    = (float*)carve((size_t)NPAD * HID * 4);
  float* agg   = (float*)carve((size_t)NPAD * HID * 4);
  float* dco   = (float*)carve((size_t)NN * 4);
  float* dre   = (float*)carve((size_t)NN * 4);
  bf16*  wst   = (bf16*) carve((size_t)31 * HH * 2);   // [0..2]=coT [3..5]=reT [6]=pnW [7..18]=zw [19..30]=zwp
  float* accco = (float*)carve((size_t)TQ * HID * 4);
  float* accre = (float*)carve((size_t)TQ * HID * 4);
  float* pns   = (float*)carve((size_t)CR * HID * 4);
  bf16*  tokco = (bf16*) carve((size_t)TQ * HID * 2);
  bf16*  tokre = (bf16*) carve((size_t)TQ * HID * 2);
  bf16*  tokpn = (bf16*) carve((size_t)TQ * HID * 2);

  const int TPB = 256;
  const long NH = (long)NPAD * HID;
  // ---- staging ----
  k_stage_pad   <<<(NH + TPB - 1) / TPB, TPB, 0, stream>>>(wordb, word, (long)NN * HID, NH);
  k_stage_T     <<<dim3(HH / TPB, 1, 3), TPB, 0, stream>>>(wst, co_W);
  k_stage_T     <<<dim3(HH / TPB, 1, 3), TPB, 0, stream>>>(wst + (size_t)3 * HH, re_W);
  k_stage_direct<<<(HH + TPB - 1) / TPB, TPB, 0, stream>>>(wst + (size_t)6 * HH, pn_W, (long)HH);
  k_stage_direct<<<((long)12 * HH + TPB - 1) / TPB, TPB, 0, stream>>>(wst + (size_t)7 * HH, zero_w, (long)12 * HH);
  k_stage_direct<<<((long)12 * HH + TPB - 1) / TPB, TPB, 0, stream>>>(wst + (size_t)19 * HH, zero_w_pos, (long)12 * HH);
  k_zero_bf16   <<<(((long)(NPAD - NN) * HID) + TPB - 1) / TPB, TPB, 0, stream>>>(hb + (size_t)NN * HID, (long)(NPAD - NN) * HID);
  // ---- degrees ----
  k_zero_f32<<<(NN + TPB - 1) / TPB, TPB, 0, stream>>>(dco, NN);
  k_zero_f32<<<(NN + TPB - 1) / TPB, TPB, 0, stream>>>(dre, NN);
  k_deg <<<(EDG + TPB - 1) / TPB, TPB, 0, stream>>>(dco, co_ei + EDG, co_ew, EDG);
  k_deg <<<(EDG + TPB - 1) / TPB, TPB, 0, stream>>>(dre, re_ei + EDG, re_ew, EDG);
  k_dinv<<<(NN + TPB - 1) / TPB, TPB, 0, stream>>>(dco, NN);
  k_dinv<<<(NN + TPB - 1) / TPB, TPB, 0, stream>>>(dre, NN);

  const int MPAIRS = NPAD / 32;                        // 954
  const dim3 gemmGrid((MPAIRS + 7) / 8, HID / 64);     // (120, 12)
  auto gcn = [&](const bf16* wT, const float* bias, const int* ei, const float* ew,
                 float* dinv, float* acc) {
    k_zero_f32<<<((long)TQ * HID + TPB - 1) / TPB, TPB, 0, stream>>>(acc, (long)TQ * HID);
    const bf16* hsrc = wordb;
    for (int i = 0; i < 3; i++) {
      k_gemm2   <<<gemmGrid, TPB, SHB, stream>>>(hsrc, wT + (size_t)i * HH, hW, MPAIRS);
      k_zero_f32<<<(NH + TPB - 1) / TPB, TPB, 0, stream>>>(agg, NH);
      k_scatter <<<EDG / EPB, 192, 0, stream>>>(agg, hW, ei, ei + EDG, ew, dinv);
      k_finalize<<<dim3(HID / TPB, NN), TPB, 0, stream>>>(hW, agg, dinv, bias + i * HID, hb);
      k_gather  <<<dim3(HID / TPB, TQ), TPB, 0, stream>>>(acc, hW, input_ids);
      hsrc = hb;
    }
  };
  gcn(wst,                  co_b, co_ei, co_ew, dco, accco);
  gcn(wst + (size_t)3 * HH, re_b, re_ei, re_ew, dre, accre);

  k_tok<<<((long)TQ * HID + TPB - 1) / TPB, TPB, 0, stream>>>(tokco, accco, 1.f / 3.f, (long)TQ * HID);
  k_tok<<<((long)TQ * HID + TPB - 1) / TPB, TPB, 0, stream>>>(tokre, accre, 1.f / 3.f, (long)TQ * HID);
  // ---- pos-neg codebook ----
  k_gemm2    <<<gemmGrid, TPB, SHB, stream>>>(wordb, wst + (size_t)6 * HH, hW, MPAIRS);
  k_pn_small <<<dim3(HID / TPB, CR), TPB, 0, stream>>>(pns, pn_graph, hW, pn_b);
  k_pn_gather<<<dim3(HID / TPB, TQ), TPB, 0, stream>>>(tokpn, pns, class_ids);
  // ---- zero-convs (blocked WMMA, fused epilogue) ----
  float* out = (float*)d_out;
  const long OUTSZ = (long)LQ * BQ * (SQ + 1) * HID;
  const dim3 zcGrid((TQ / 32) / 8, HID / 64, LQ);      // (8, 12, 12)
  k_zeroconv2<<<zcGrid, TPB, SHB, stream>>>(tokco, wst + (size_t)7 * HH,  zero_b,     mask,    out);
  k_zeroconv2<<<zcGrid, TPB, SHB, stream>>>(tokre, wst + (size_t)7 * HH,  zero_b,     mask,    out + OUTSZ);
  k_zeroconv2<<<zcGrid, TPB, SHB, stream>>>(tokpn, wst + (size_t)19 * HH, zero_b_pos, nullptr, out + 2 * OUTSZ);
  k_outpad  <<<(3 * LQ * BQ * HID + TPB - 1) / TPB, TPB, 0, stream>>>(out);
}
